// LocalFeatureExtractorLayer_70875550318862
// MI455X (gfx1250) — matrix operand order repaired and checked by reading, not compiled, not run on hardware
//
#include <hip/hip_runtime.h>

typedef __attribute__((ext_vector_type(16))) _Float16 v16h;
typedef __attribute__((ext_vector_type(8)))  _Float16 v8h;
typedef __attribute__((ext_vector_type(8)))  float    v8f;

#define BB  8
#define NN  1024
#define KNN 20

// ---------------------------------------------------------------------------
// WMMA fragment helpers (v_wmma_f32_16x16x32_f16, wave32)
// A layout (16x32 f16, doc 7.12.2): lane lo -> M row; element e<8 -> K = kk+hi*8+e,
// e>=8 -> K = kk+hi*8+16+(e-8)  => two contiguous 8-half (16B) runs per lane.
// B layout (32x16 f16): lane lo -> N col; same K order as A:
// e<8 -> K = kk+hi*8+e, e>=8 -> K = kk+hi*8+16+(e-8)  => two 8-half runs per lane.
// ---------------------------------------------------------------------------
static __device__ inline v16h frag_from_runs(v8h r0, v8h r1) {
  v16h a;
#pragma unroll
  for (int i = 0; i < 8; i++) { a[i] = r0[i]; a[i + 8] = r1[i]; }
  return a;
}

static __device__ inline v16h load_fragA(const _Float16* __restrict__ row, int kk, int hi) {
  v8h r0 = *(const v8h*)(row + kk + hi * 8);
  v8h r1 = *(const v8h*)(row + kk + hi * 8 + 16);
  return frag_from_runs(r0, r1);
}

static __device__ inline v16h load_fragB(const _Float16* __restrict__ col, int kk, int hi) {
  v8h r0 = *(const v8h*)(col + kk + hi * 8);
  v8h r1 = *(const v8h*)(col + kk + hi * 8 + 16);
  return frag_from_runs(r0, r1);
}

// Edge A fragment with [neighbor | center] concat; C is a multiple of 16 so
// each 8-half run lies entirely on one side.
static __device__ inline v16h load_fragA_edge(const _Float16* __restrict__ nbrRow,
                                              const _Float16* __restrict__ ctrRow,
                                              int C, int kk, int hi) {
  int k0 = kk + hi * 8, k1 = k0 + 16;
  const _Float16* p0 = (k0 < C) ? (nbrRow + k0) : (ctrRow + (k0 - C));
  const _Float16* p1 = (k1 < C) ? (nbrRow + k1) : (ctrRow + (k1 - C));
  return frag_from_runs(*(const v8h*)p0, *(const v8h*)p1);
}

// ---------------------------------------------------------------------------
// prep: transpose input (B,10,N)->(B,N,32) f16 zero-padded, + squared norms
// ---------------------------------------------------------------------------
__global__ void prep_input_kernel(const float* __restrict__ inp,
                                  _Float16* __restrict__ Xh,
                                  float* __restrict__ sq) {
  int pt = blockIdx.x, lane = threadIdx.x;     // 32 lanes
  int b = pt >> 10, n = pt & 1023;
  float v = 0.f;
  if (lane < 10) v = inp[((size_t)b * 10 + lane) * NN + n];
  Xh[(size_t)pt * 32 + lane] = (_Float16)v;
  float s = v * v;
#pragma unroll
  for (int off = 16; off > 0; off >>= 1) s += __shfl_xor(s, off, 32);
  if (lane == 0) sq[pt] = s;
}

// ---------------------------------------------------------------------------
// prep: fold BN scale into weights -> f16; fused bias. layer1 pads 20->32
// with nbr half at [0,16) and ctr half at [16,32).
// ---------------------------------------------------------------------------
__global__ void prep_w_kernel(const float* __restrict__ w, const float* __restrict__ bb,
                              const float* __restrict__ g, const float* __restrict__ beta,
                              const float* __restrict__ m, const float* __restrict__ var,
                              _Float16* __restrict__ wh, float* __restrict__ bias2,
                              int Cw, int CwPad, int layer1) {
  int o = blockIdx.x;
  float sc = g[o] * rsqrtf(var[o] + 1e-5f);
  if (threadIdx.x == 0) bias2[o] = (bb[o] - m[o]) * sc + beta[o];
  for (int k = threadIdx.x; k < CwPad; k += blockDim.x) {
    float val = 0.f;
    if (layer1) {
      if (k < 10) val = w[o * 20 + k];
      else if (k >= 16 && k < 26) val = w[o * 20 + (k - 6)];
    } else if (k < Cw) {
      val = w[(size_t)o * Cw + k];
    }
    wh[(size_t)o * CwPad + k] = (_Float16)(val * sc);
  }
}

// ---------------------------------------------------------------------------
// Pairwise distances: D2[b][n][m] = |x_n|^2 + |x_m|^2 - 2 x_n.x_m
// One wave per 16x64 tile (4 accumulators share one A fragment), WMMA over K.
// ---------------------------------------------------------------------------
__global__ void gram_kernel(const _Float16* __restrict__ Fh, int rowStride, int kIters,
                            const float* __restrict__ sq, float* __restrict__ D2) {
  int b = blockIdx.z, n0 = blockIdx.y * 16, m0 = blockIdx.x * 64;
  int l = threadIdx.x, lo = l & 15, hi = l >> 4;
  const _Float16* base = Fh + (size_t)b * NN * rowStride;
  const _Float16* arow = base + (size_t)(n0 + lo) * rowStride;
  const _Float16* brow[4];
#pragma unroll
  for (int j = 0; j < 4; j++) brow[j] = base + (size_t)(m0 + j * 16 + lo) * rowStride;
  v8f acc[4] = {};
  for (int t = 0; t < kIters; t++) {
    int kk = t * 32;
    v16h a = load_fragA(arow, kk, hi);
#pragma unroll
    for (int j = 0; j < 4; j++) {
      v16h bf = load_fragB(brow[j], kk, hi);
      acc[j] = __builtin_amdgcn_wmma_f32_16x16x32_f16(false, a, false, bf, (short)0,
                                                      acc[j], false, false);
    }
  }
  const float* sqb = sq + b * NN;
  float* Drow = D2 + ((size_t)b * NN + n0) * NN + m0;
#pragma unroll
  for (int j = 0; j < 4; j++) {
    float sqm = sqb[m0 + j * 16 + lo];
#pragma unroll
    for (int r = 0; r < 8; r++) {
      int M = r + hi * 8;
      Drow[(size_t)M * NN + j * 16 + lo] = sqb[n0 + M] + sqm - 2.0f * acc[j][r];
    }
  }
}

// ---------------------------------------------------------------------------
// top-k (k=20) smallest per row: one wave per point, row in LDS,
// 20 rounds of wave-wide argmin, winner poisoned to +INF.
// ---------------------------------------------------------------------------
__global__ void __launch_bounds__(256) topk_kernel(const float* __restrict__ D2,
                                                   int* __restrict__ IDX) {
  __shared__ float ld[8][NN];
  int wave = threadIdx.x >> 5, lane = threadIdx.x & 31;
  int pt = blockIdx.x * 8 + wave;
  const float* row = D2 + (size_t)pt * NN;
  for (int mc = lane; mc < NN; mc += 32) ld[wave][mc] = row[mc];
  __syncthreads();
  int* out = IDX + (size_t)pt * KNN;
  for (int sel = 0; sel < KNN; sel++) {
    float best = 3.0e38f; int bi = 0;
    for (int mc = lane; mc < NN; mc += 32) {
      float v = ld[wave][mc];
      if (v < best) { best = v; bi = mc; }
    }
#pragma unroll
    for (int off = 16; off > 0; off >>= 1) {
      float ov = __shfl_xor(best, off, 32);
      int   oi = __shfl_xor(bi,   off, 32);
      if (ov < best || (ov == best && oi < bi)) { best = ov; bi = oi; }
    }
    if (lane == 0) { out[sel] = bi; ld[wave][bi] = 3.0e38f; }
    __syncthreads();
  }
}

__global__ void zero_kernel(float* __restrict__ p, int count) {
  int i = blockIdx.x * blockDim.x + threadIdx.x;
  if (i < count) p[i] = 0.f;
}

// ---------------------------------------------------------------------------
// EdgeConv GEMM: rows = (point, neighbor) pairs (N*20 per batch).
// A = concat[nbr, ctr] f16 gathered per row; B = folded W^T.
// 16x64 output tile per wave: gathered A fragment reused across 4 WMMAs.
// Epilogue: bias + ReLU, then float-bit atomicMax into (B,N,O) (max over k).
// ---------------------------------------------------------------------------
__global__ void edgeconv_kernel(const _Float16* __restrict__ Fh, int rowStride, int C,
                                const int* __restrict__ IDX,
                                const _Float16* __restrict__ Wh,
                                const float* __restrict__ bias2,
                                float* __restrict__ out32, int O) {
  int b = blockIdx.z, rowTile = blockIdx.y, o0 = blockIdx.x * 64;
  int l = threadIdx.x, lo = l & 15, hi = l >> 4;
  int row = rowTile * 16 + lo;
  int n = row / KNN, kn = row % KNN;
  int nbr = IDX[((size_t)b * NN + n) * KNN + kn];
  const _Float16* fb = Fh + (size_t)b * NN * rowStride;
  const _Float16* nbrRow = fb + (size_t)nbr * rowStride;
  const _Float16* ctrRow = fb + (size_t)n * rowStride;
  const _Float16* wRow[4];
#pragma unroll
  for (int j = 0; j < 4; j++) wRow[j] = Wh + (size_t)(o0 + j * 16 + lo) * (2 * C);
  v8f acc[4] = {};
  int kIters = (2 * C) / 32;
  for (int t = 0; t < kIters; t++) {
    int kk = t * 32;
    v16h a = load_fragA_edge(nbrRow, ctrRow, C, kk, hi);
#pragma unroll
    for (int j = 0; j < 4; j++) {
      v16h bf = load_fragB(wRow[j], kk, hi);
      acc[j] = __builtin_amdgcn_wmma_f32_16x16x32_f16(false, a, false, bf, (short)0,
                                                      acc[j], false, false);
    }
  }
#pragma unroll
  for (int j = 0; j < 4; j++) {
    int o = o0 + j * 16 + lo;
    float bias = bias2[o];
#pragma unroll
    for (int r = 0; r < 8; r++) {
      int row2 = rowTile * 16 + r + hi * 8;
      int n2 = row2 / KNN;
      float val = acc[j][r] + bias;
      val = val > 0.f ? val : 0.f;  // ReLU >= 0 -> int-bit compare is order-preserving
      atomicMax((int*)&out32[((size_t)b * NN + n2) * O + o], __float_as_int(val));
    }
  }
}

// ---------------------------------------------------------------------------
// f32 layer output -> f16 slice of concatenated feature buffer + row norms
// ---------------------------------------------------------------------------
__global__ void finalize_kernel(const float* __restrict__ in32,
                                _Float16* __restrict__ outSlice,
                                float* __restrict__ sq, int O) {
  int pt = blockIdx.x, lane = threadIdx.x;
  const float* r = in32 + (size_t)pt * O;
  _Float16* wrow = outSlice + (size_t)pt * 1024;
  float s = 0.f;
  for (int c = lane; c < O; c += 32) {
    float v = r[c];
    s += v * v;
    wrow[c] = (_Float16)v;
  }
#pragma unroll
  for (int off = 16; off > 0; off >>= 1) s += __shfl_xor(s, off, 32);
  if (lane == 0) sq[pt] = s;
}

// ---------------------------------------------------------------------------
// Final GEMM: (B*N,1024) x (1024,1024)^T, bias+ReLU, output transposed (B,1024,N).
// 16x64 output tile per wave.
// ---------------------------------------------------------------------------
__global__ void gemm5_kernel(const _Float16* __restrict__ FhAll,
                             const _Float16* __restrict__ Wh,
                             const float* __restrict__ bias2,
                             float* __restrict__ out) {
  int b = blockIdx.z, n0 = blockIdx.y * 16, o0 = blockIdx.x * 64;
  int l = threadIdx.x, lo = l & 15, hi = l >> 4;
  const _Float16* arow = FhAll + ((size_t)b * NN + n0 + lo) * 1024;
  const _Float16* wrow[4];
#pragma unroll
  for (int j = 0; j < 4; j++) wrow[j] = Wh + (size_t)(o0 + j * 16 + lo) * 1024;
  v8f acc[4] = {};
  for (int t = 0; t < 32; t++) {
    int kk = t * 32;
    v16h a = load_fragA(arow, kk, hi);
#pragma unroll
    for (int j = 0; j < 4; j++) {
      v16h bf = load_fragB(wrow[j], kk, hi);
      acc[j] = __builtin_amdgcn_wmma_f32_16x16x32_f16(false, a, false, bf, (short)0,
                                                      acc[j], false, false);
    }
  }
#pragma unroll
  for (int j = 0; j < 4; j++) {
    int o = o0 + j * 16 + lo;
    float bias = bias2[o];
    float* orow = out + ((size_t)b * 1024 + o) * NN + n0 + hi * 8;
    float4 s0, s1;
    s0.x = fmaxf(acc[j][0] + bias, 0.f); s0.y = fmaxf(acc[j][1] + bias, 0.f);
    s0.z = fmaxf(acc[j][2] + bias, 0.f); s0.w = fmaxf(acc[j][3] + bias, 0.f);
    s1.x = fmaxf(acc[j][4] + bias, 0.f); s1.y = fmaxf(acc[j][5] + bias, 0.f);
    s1.z = fmaxf(acc[j][6] + bias, 0.f); s1.w = fmaxf(acc[j][7] + bias, 0.f);
    *(float4*)(orow)     = s0;
    *(float4*)(orow + 4) = s1;
  }
}

// ---------------------------------------------------------------------------
extern "C" void kernel_launch(void* const* d_in, const int* in_sizes, int n_in,
                              void* d_out, int out_size, void* d_ws, size_t ws_size,
                              hipStream_t stream) {
  (void)in_sizes; (void)n_in; (void)out_size; (void)ws_size;
  const float* input = (const float*)d_in[0];
  const float* p[5][6];
  for (int L = 0; L < 5; L++)
    for (int j = 0; j < 6; j++) p[L][j] = (const float*)d_in[1 + L * 6 + j];

  char* wsb = (char*)d_ws;
  size_t off = 0;
  auto carve = [&](size_t bytes) -> char* {
    char* r = wsb + off;
    off += (bytes + 255) & ~(size_t)255;
    return r;
  };
  _Float16* Xh    = (_Float16*)carve((size_t)BB * NN * 32 * 2);
  _Float16* FhAll = (_Float16*)carve((size_t)BB * NN * 1024 * 2);
  float*    sq    = (float*)carve((size_t)BB * NN * 4);
  float*    D2    = (float*)carve((size_t)BB * NN * NN * 4);
  int*      IDX   = (int*)carve((size_t)BB * NN * KNN * 4);
  float*    F32   = (float*)carve((size_t)BB * NN * 512 * 4);
  _Float16* Wh1   = (_Float16*)carve((size_t)128 * 32 * 2);
  _Float16* Wh2   = (_Float16*)carve((size_t)128 * 256 * 2);
  _Float16* Wh3   = (_Float16*)carve((size_t)256 * 256 * 2);
  _Float16* Wh4   = (_Float16*)carve((size_t)512 * 512 * 2);
  _Float16* Wh5   = (_Float16*)carve((size_t)1024 * 1024 * 2);
  float* bi1 = (float*)carve(128 * 4);
  float* bi2 = (float*)carve(128 * 4);
  float* bi3 = (float*)carve(256 * 4);
  float* bi4 = (float*)carve(512 * 4);
  float* bi5 = (float*)carve(1024 * 4);

  prep_input_kernel<<<BB * NN, 32, 0, stream>>>(input, Xh, sq);
  prep_w_kernel<<<128, 64, 0, stream>>>(p[0][0], p[0][1], p[0][2], p[0][3], p[0][4], p[0][5], Wh1, bi1, 20, 32, 1);
  prep_w_kernel<<<128, 64, 0, stream>>>(p[1][0], p[1][1], p[1][2], p[1][3], p[1][4], p[1][5], Wh2, bi2, 256, 256, 0);
  prep_w_kernel<<<256, 64, 0, stream>>>(p[2][0], p[2][1], p[2][2], p[2][3], p[2][4], p[2][5], Wh3, bi3, 256, 256, 0);
  prep_w_kernel<<<512, 64, 0, stream>>>(p[3][0], p[3][1], p[3][2], p[3][3], p[3][4], p[3][5], Wh4, bi4, 512, 512, 0);
  prep_w_kernel<<<1024, 64, 0, stream>>>(p[4][0], p[4][1], p[4][2], p[4][3], p[4][4], p[4][5], Wh5, bi5, 1024, 1024, 0);

  dim3 gGram(NN / 64, NN / 16, BB);
  dim3 gEdge;
  int topkBlocks = (BB * NN) / 8;

  // ----- Layer 1: x(10, padded 16) -> f1(128) -----
  gram_kernel<<<gGram, 32, 0, stream>>>(Xh, 32, 1, sq, D2);
  topk_kernel<<<topkBlocks, 256, 0, stream>>>(D2, IDX);
  zero_kernel<<<(BB * NN * 128 + 255) / 256, 256, 0, stream>>>(F32, BB * NN * 128);
  gEdge = dim3(128 / 64, (NN * KNN) / 16, BB);
  edgeconv_kernel<<<gEdge, 32, 0, stream>>>(Xh, 32, 16, IDX, Wh1, bi1, F32, 128);
  finalize_kernel<<<BB * NN, 32, 0, stream>>>(F32, FhAll + 0, sq, 128);

  // ----- Layer 2: f1(128) -> f2(128) -----
  gram_kernel<<<gGram, 32, 0, stream>>>(FhAll + 0, 1024, 4, sq, D2);
  topk_kernel<<<topkBlocks, 256, 0, stream>>>(D2, IDX);
  zero_kernel<<<(BB * NN * 128 + 255) / 256, 256, 0, stream>>>(F32, BB * NN * 128);
  edgeconv_kernel<<<gEdge, 32, 0, stream>>>(FhAll + 0, 1024, 128, IDX, Wh2, bi2, F32, 128);
  finalize_kernel<<<BB * NN, 32, 0, stream>>>(F32, FhAll + 128, sq, 128);

  // ----- Layer 3: f2(128) -> f3(256) -----
  gram_kernel<<<gGram, 32, 0, stream>>>(FhAll + 128, 1024, 4, sq, D2);
  topk_kernel<<<topkBlocks, 256, 0, stream>>>(D2, IDX);
  zero_kernel<<<(BB * NN * 256 + 255) / 256, 256, 0, stream>>>(F32, BB * NN * 256);
  gEdge = dim3(256 / 64, (NN * KNN) / 16, BB);
  edgeconv_kernel<<<gEdge, 32, 0, stream>>>(FhAll + 128, 1024, 128, IDX, Wh3, bi3, F32, 256);
  finalize_kernel<<<BB * NN, 32, 0, stream>>>(F32, FhAll + 256, sq, 256);

  // ----- Layer 4: f3(256) -> f4(512) -----
  gram_kernel<<<gGram, 32, 0, stream>>>(FhAll + 256, 1024, 8, sq, D2);
  topk_kernel<<<topkBlocks, 256, 0, stream>>>(D2, IDX);
  zero_kernel<<<(BB * NN * 512 + 255) / 256, 256, 0, stream>>>(F32, BB * NN * 512);
  gEdge = dim3(512 / 64, (NN * KNN) / 16, BB);
  edgeconv_kernel<<<gEdge, 32, 0, stream>>>(FhAll + 256, 1024, 256, IDX, Wh4, bi4, F32, 512);
  finalize_kernel<<<BB * NN, 32, 0, stream>>>(F32, FhAll + 512, sq, 512);

  // ----- Layer 5: concat(1024) -> out (B,1024,N) -----
  gemm5_kernel<<<dim3(1024 / 64, NN / 16, BB), 32, 0, stream>>>(FhAll, Wh5, bi5, (float*)d_out);
}
